// QuantLinear_54408645705773
// MI455X (gfx1250) — matrix operand order repaired and checked by reading, not compiled, not run on hardware
//
#include <hip/hip_runtime.h>
#include <stdint.h>

typedef __attribute__((ext_vector_type(16))) _Float16 v16h;
typedef __attribute__((ext_vector_type(8)))  _Float16 v8h;
typedef __attribute__((ext_vector_type(2)))  _Float16 h2;
typedef __attribute__((ext_vector_type(8)))  float    v8f;

#define MT 8   // 16x16 M-tiles per wave -> 128 rows   (amortizes dequant)
#define NT 2   // 16x16 N-tiles per wave -> 32 cols
// 8 waves per block, 2 (M) x 4 (N) -> block tile 256 x 128

union AU { v16h v; struct { v8h lo, hi; } s; };
union BU { v16h v; h2 p[8]; };

__global__ __launch_bounds__(256) void qlin_w4a16_wmma(
    const _Float16* __restrict__ x,        // (M, K) fp16
    const uint32_t* __restrict__ qweight,  // (K/8, N) packed nibbles along K
    const _Float16* __restrict__ scales,   // (N,) fp16
    const uint32_t* __restrict__ qzeros,   // (N/8,) packed nibbles along N
    const _Float16* __restrict__ bias,     // (N,) fp16
    _Float16* __restrict__ y,              // (M, N) fp16
    int M, int N, int K)
{
    const int lane = threadIdx.x & 31;
    const int wave = threadIdx.x >> 5;
    const int hi   = lane >> 4;     // half-wave select (0/1)
    const int ln   = lane & 15;
    const int waveM = wave >> 2;    // 0..1
    const int waveN = wave & 3;     // 0..3
    const int m0 = blockIdx.y * 256 + waveM * (MT * 16);
    const int n0 = blockIdx.x * 128 + waveN * (NT * 16);

    // ---- per-column dequant constants (fixed over the whole K loop) ----
    // W[k][n] = w4*s - (z4+1)*s.  Raw half via 0x6400 trick is (1024 + w4),
    // so W = fma(raw, s, -(1024 + z4 + 1)*s).
    h2 sc2[NT], off2[NT];
#pragma unroll
    for (int nt = 0; nt < NT; ++nt) {
        int n = n0 + nt * 16 + ln;
        float s = (float)scales[n];
        int z = (int)((qzeros[n >> 3] >> ((n & 7) * 4)) & 15u);
        float c = -(1024.0f + (float)(z + 1)) * s;
        sc2[nt][0]  = (_Float16)s;  sc2[nt][1]  = (_Float16)s;
        off2[nt][0] = (_Float16)c;  off2[nt][1] = (_Float16)c;
    }

    v8f acc[MT][NT];
#pragma unroll
    for (int mi = 0; mi < MT; ++mi)
#pragma unroll
        for (int nt = 0; nt < NT; ++nt)
            acc[mi][nt] = (v8f){0.f, 0.f, 0.f, 0.f, 0.f, 0.f, 0.f, 0.f};

    const int KT = K >> 5;               // K-steps of 32
    const int koffA = 16 * hi;           // A half-wave K offset

    for (int kt = 0; kt < KT; ++kt) {
        const int kb = kt << 5;
        const int r0 = (kb >> 3) + 2 * hi;

        if (kt + 1 < KT) {
            __builtin_prefetch(x + (size_t)(m0 + ln) * K + kb + 32, 0, 0);
            __builtin_prefetch(&qweight[(size_t)(r0 + 4) * N + n0 + ln], 0, 0);
        }

        // ---- issue all loads for this K-step up front ----
        uint32_t qd[NT][2];
#pragma unroll
        for (int nt = 0; nt < NT; ++nt) {
            const int n = n0 + nt * 16 + ln;
            qd[nt][0] = qweight[(size_t)r0 * N + n];
            qd[nt][1] = qweight[(size_t)(r0 + 1) * N + n];
        }

        // A fragments: lane l holds row (l&15); halves 0-15 = K kb+16*hi..+15
        // (contiguous, matching the B dequant order)
        AU a[MT];
#pragma unroll
        for (int mi = 0; mi < MT; ++mi) {
            const _Float16* xp = x + (size_t)(m0 + mi * 16 + ln) * K + kb + koffA;
            a[mi].s.lo = *(const v8h*)(xp);
            a[mi].s.hi = *(const v8h*)(xp + 8);
        }

        // ---- per N-tile: dequant 16 halves (v_perm + or + pk_fma) + 8 WMMAs ----
#pragma unroll
        for (int nt = 0; nt < NT; ++nt) {
            const uint32_t q0 = qd[nt][0], q1 = qd[nt][1];
            const uint32_t ql0 = q0 & 0x0F0F0F0Fu, qh0 = (q0 >> 4) & 0x0F0F0F0Fu;
            const uint32_t ql1 = q1 & 0x0F0F0F0Fu, qh1 = (q1 >> 4) & 0x0F0F0F0Fu;
            BU bu;
#pragma unroll
            for (int p = 0; p < 4; ++p) {
                // dest bytes: {ql.byte[p], 0x00, qh.byte[p], 0x00} then |0x64006400
                const uint32_t sel = 0x0C040C00u + (uint32_t)p * 0x00010001u;
                uint32_t t0 = __builtin_amdgcn_perm(qh0, ql0, sel) | 0x64006400u;
                uint32_t t1 = __builtin_amdgcn_perm(qh1, ql1, sel) | 0x64006400u;
                bu.p[p]     = __builtin_bit_cast(h2, t0) * sc2[nt] + off2[nt];
                bu.p[4 + p] = __builtin_bit_cast(h2, t1) * sc2[nt] + off2[nt];
            }
#pragma unroll
            for (int mi = 0; mi < MT; ++mi) {
                acc[mi][nt] = __builtin_amdgcn_wmma_f32_16x16x32_f16(
                    /*neg_a=*/false, a[mi].v, /*neg_b=*/false, bu.v,
                    /*c_mod=*/(short)0, acc[mi][nt],
                    /*reuse_a=*/false, /*reuse_b=*/false);
            }
        }
    }

    // ---- epilogue: C/D layout -> M = v + 8*hi, N = l&15 per 16x16 tile.
    //      fp32->fp16 cast, packed fp16 bias add (matches reference). ----
#pragma unroll
    for (int nt = 0; nt < NT; ++nt) {
        const int n = n0 + nt * 16 + ln;
        const _Float16 bh = bias[n];
        h2 bh2; bh2[0] = bh; bh2[1] = bh;
#pragma unroll
        for (int mi = 0; mi < MT; ++mi) {
            const int mb = m0 + mi * 16 + 8 * hi;
#pragma unroll
            for (int v = 0; v < 8; v += 2) {
                h2 o;
                o[0] = (_Float16)acc[mi][nt][v];
                o[1] = (_Float16)acc[mi][nt][v + 1];
                o = o + bh2;                       // v_pk_add_f16
                y[(size_t)(mb + v)     * N + n] = o[0];
                y[(size_t)(mb + v + 1) * N + n] = o[1];
            }
        }
    }
}

extern "C" void kernel_launch(void* const* d_in, const int* in_sizes, int n_in,
                              void* d_out, int out_size, void* d_ws, size_t ws_size,
                              hipStream_t stream) {
    const _Float16* x       = (const _Float16*)d_in[0];
    const uint32_t* qweight = (const uint32_t*)d_in[1];
    const _Float16* scales  = (const _Float16*)d_in[2];
    const uint32_t* qzeros  = (const uint32_t*)d_in[3];
    const _Float16* bias    = (const _Float16*)d_in[4];
    _Float16* y = (_Float16*)d_out;

    const int N = in_sizes[4];                                  // bias length
    const int K = (int)(((long long)in_sizes[1] * 8) / N);      // qweight = (K/8, N)
    const int M = (int)((long long)in_sizes[0] / K);            // x = (M, K)

    dim3 grid((unsigned)(N / 128), (unsigned)(M / 256));
    dim3 block(256);
    qlin_w4a16_wmma<<<grid, block, 0, stream>>>(x, qweight, scales, qzeros, bias, y, M, N, K);
}
